// DimensionScaledEqProp_46084999086240
// MI455X (gfx1250) — matrix operand order repaired and checked by reading, not compiled, not run on hardware
//
#include <hip/hip_runtime.h>
#include <hip/hip_bf16.h>
#include <math.h>

// ---------------------------------------------------------------------------
// Problem constants (from reference)
// ---------------------------------------------------------------------------
#define BDIM   4096
#define DIN    512
#define DH     1024
#define DOUT   256
#define DFF    4096
#define STEPS  30
#define LN_EPS 1e-5f
#define GAMMA_C 0.17677669529663687f   // 0.5*sqrt(64/512)

// ---------------------------------------------------------------------------
// Types for WMMA
// ---------------------------------------------------------------------------
typedef __bf16 bf16;
typedef bf16  bf16x8 __attribute__((ext_vector_type(8)));
typedef bf16  v16bf  __attribute__((ext_vector_type(16)));
typedef float v8f    __attribute__((ext_vector_type(8)));

// GEMM tiling: block = 256 threads = 8 waves (wave32).
// Macro tile 128x128, K-step 32. Waves arranged 4(M) x 2(N);
// each wave computes a 32x64 tile via 2x4 accumulators of v_wmma_f32_16x16x32_bf16.
// Global->LDS staging uses gfx1250 async-DMA (global_load_async_to_lds_b128),
// double-buffered and tracked with ASYNCcnt.
#define BM 128
#define BN 128
#define BK 32

enum { EPI_BIAS_F32 = 0, EPI_TANH_BF16 = 1, EPI_HUPDATE = 2 };

// Issue one 16-byte async DMA chunk: global -> LDS. lds_addr is the
// workgroup-relative LDS byte offset (= low 32 bits of the flat pointer).
__device__ __forceinline__ void async_b128(uint32_t lds_addr, uint64_t gaddr) {
  asm volatile("global_load_async_to_lds_b128 %0, %1, off"
               :: "v"(lds_addr), "v"(gaddr) : "memory");
}

// C[M,N] = A[M,K] * B[N,K]^T  (both bf16 row-major, f32 accumulate)
template <int MODE>
__global__ __launch_bounds__(256) void gemm_bt(
    const bf16* __restrict__ A, const bf16* __restrict__ Bm,
    const float* __restrict__ bias,
    float* __restrict__ outF, bf16* __restrict__ outB,
    const float* __restrict__ xemb, float* __restrict__ hio,
    int M, int N, int K)
{
  __shared__ __align__(16) bf16 As[2][BM * BK];
  __shared__ __align__(16) bf16 Bs[2][BN * BK];

  const int tid  = threadIdx.x;
  const int lane = tid & 31;
  const int wave = tid >> 5;
  const int lr   = lane & 15;   // row/col index within 16-wide fragment
  const int hs   = lane >> 4;   // half-select (lane group 0-15 vs 16-31)
  const int wm   = (wave & 3) * 32;   // wave M offset in macro tile
  const int wn   = (wave >> 2) * 64;  // wave N offset in macro tile

  const int tileN = blockIdx.x * BN;
  const int tileM = blockIdx.y * BM;

  // ---- per-thread DMA chunk addresses (2 chunks of A + 2 of B per thread;
  //      512 x 16B chunks cover each 128x32 bf16 tile)
  uint64_t aG[2], bG[2];
  uint32_t aL[2], bL[2];
#pragma unroll
  for (int c = 0; c < 2; ++c) {
    const int idx = tid + c * 256;       // 0..511
    const int row = idx >> 2;            // 0..127
    const int ck  = (idx & 3) * 8;       // element offset 0,8,16,24
    aG[c] = (uint64_t)(uintptr_t)(A  + (size_t)(tileM + row) * K + ck);
    bG[c] = (uint64_t)(uintptr_t)(Bm + (size_t)(tileN + row) * K + ck);
    aL[c] = (uint32_t)((row * BK + ck) * sizeof(bf16));
    bL[c] = (uint32_t)((row * BK + ck) * sizeof(bf16));
  }
  const uint32_t asBase = (uint32_t)(uintptr_t)&As[0][0];
  const uint32_t bsBase = (uint32_t)(uintptr_t)&Bs[0][0];
  const uint32_t BUFB   = (uint32_t)(BM * BK * sizeof(bf16));  // 8 KB

  auto issue_tile = [&](int kt, int buf) {
    const uint64_t koff = (uint64_t)kt * (BK * sizeof(bf16));
#pragma unroll
    for (int c = 0; c < 2; ++c) {
      async_b128(asBase + buf * BUFB + aL[c], aG[c] + koff);
      async_b128(bsBase + buf * BUFB + bL[c], bG[c] + koff);
    }
  };

  v8f acc[2][4];
  const v8f vzero = {0.f, 0.f, 0.f, 0.f, 0.f, 0.f, 0.f, 0.f};
#pragma unroll
  for (int i = 0; i < 2; ++i)
#pragma unroll
    for (int j = 0; j < 4; ++j) acc[i][j] = vzero;

  const int nK = K / BK;
  issue_tile(0, 0);                      // prologue: tile 0 -> buffer 0

  for (int kt = 0; kt < nK; ++kt) {
    const int cur = kt & 1;
    if (kt + 1 < nK) {
      issue_tile(kt + 1, cur ^ 1);       // overlap next tile's DMA with WMMA
      asm volatile("s_wait_asynccnt 0x4" ::: "memory");  // tile kt's 4 chunks done
    } else {
      asm volatile("s_wait_asynccnt 0x0" ::: "memory");
    }
    __syncthreads();                     // all waves' DMA for tile kt visible

    const bf16* Abuf = &As[cur][0];
    const bf16* Bbuf = &Bs[cur][0];

    // ---- LDS -> fragments per ISA 7.12.2 (16-bit A 16x32, B 32x16 layouts)
    union Frag { v16bf v; bf16x8 h[2]; };
    Frag a[2], b[4];
#pragma unroll
    for (int i = 0; i < 2; ++i) {
      const int r = wm + i * 16 + lr;
      // lane<16: K0-7 then K16-23 ; lane>=16: K8-15 then K24-31
      a[i].h[0] = *(const bf16x8*)&Abuf[r * BK + hs * 8];
      a[i].h[1] = *(const bf16x8*)&Abuf[r * BK + 16 + hs * 8];
    }
#pragma unroll
    for (int j = 0; j < 4; ++j) {
      const int cc = wn + j * 16 + lr;
      // lane<16: K0-7 then K16-23 ; lane>=16: K8-15 then K24-31
      b[j].h[0] = *(const bf16x8*)&Bbuf[cc * BK + hs * 8];
      b[j].h[1] = *(const bf16x8*)&Bbuf[cc * BK + 16 + hs * 8];
    }

#pragma unroll
    for (int i = 0; i < 2; ++i)
#pragma unroll
      for (int j = 0; j < 4; ++j)
        acc[i][j] = __builtin_amdgcn_wmma_f32_16x16x32_bf16(
            false, a[i].v, false, b[j].v, (short)0, acc[i][j], false, false);

    __syncthreads();                     // reads done before buffer reuse
  }

  // ---- epilogue. C/D layout: VGPR e, lanes 0-15 -> M=e, N=lane;
  //      lanes 16-31 -> M=e+8, N=lane-16.
#pragma unroll
  for (int i = 0; i < 2; ++i) {
#pragma unroll
    for (int j = 0; j < 4; ++j) {
      const int rbase = tileM + wm + i * 16 + hs * 8;
      const int cbase = tileN + wn + j * 16 + lr;
      const float bv = bias[cbase];
#pragma unroll
      for (int e = 0; e < 8; ++e) {
        const size_t off = (size_t)(rbase + e) * N + cbase;
        const float val = acc[i][j][e] + bv;
        if (MODE == EPI_BIAS_F32) {
          outF[off] = val;
        } else if (MODE == EPI_TANH_BF16) {
          outB[off] = (bf16)tanhf(val);
        } else {  // EPI_HUPDATE: h = (1-g)*h + g*(val + x_emb)
          const float tgt = val + xemb[off];
          hio[off] = (1.0f - GAMMA_C) * hio[off] + GAMMA_C * tgt;
        }
      }
    }
  }
}

// ---------------------------------------------------------------------------
// LayerNorm -> bf16 (one block of 256 threads per row)
// ---------------------------------------------------------------------------
__global__ __launch_bounds__(256) void ln_bf16_kernel(
    const float* __restrict__ h, const float* __restrict__ g,
    const float* __restrict__ bb, bf16* __restrict__ out, int D)
{
  __shared__ float red[256];
  const int row = blockIdx.x;
  const float* hr = h + (size_t)row * D;
  float s = 0.f, s2 = 0.f;
  for (int c = threadIdx.x; c < D; c += 256) {
    const float x = hr[c];
    s += x; s2 += x * x;
  }
  red[threadIdx.x] = s; __syncthreads();
  for (int o = 128; o > 0; o >>= 1) {
    if (threadIdx.x < o) red[threadIdx.x] += red[threadIdx.x + o];
    __syncthreads();
  }
  const float mu = red[0] / D;
  __syncthreads();
  red[threadIdx.x] = s2; __syncthreads();
  for (int o = 128; o > 0; o >>= 1) {
    if (threadIdx.x < o) red[threadIdx.x] += red[threadIdx.x + o];
    __syncthreads();
  }
  const float var = red[0] / D - mu * mu;
  const float rstd = rsqrtf(var + LN_EPS);
  for (int c = threadIdx.x; c < D; c += 256)
    out[(size_t)row * D + c] = (bf16)((hr[c] - mu) * rstd * g[c] + bb[c]);
}

// ---------------------------------------------------------------------------
// Spectral-norm helpers (tiny f32 work)
// ---------------------------------------------------------------------------
__global__ void fill_kernel(float* __restrict__ x, float val, int n) {
  const int i = blockIdx.x * 256 + threadIdx.x;
  if (i < n) x[i] = val;
}

// u[r] = sum_c W[r,c] * v[c]   (8 rows per block, one wave per row)
__global__ __launch_bounds__(256) void mv_rows(
    const float* __restrict__ W, const float* __restrict__ v,
    float* __restrict__ u, int R, int C)
{
  const int wave = threadIdx.x >> 5, lane = threadIdx.x & 31;
  const int row = blockIdx.x * 8 + wave;
  if (row >= R) return;
  const float* wr = W + (size_t)row * C;
  float acc = 0.f;
  for (int c = lane; c < C; c += 32) acc += wr[c] * v[c];
  for (int o = 16; o > 0; o >>= 1) acc += __shfl_down(acc, o, 32);
  if (lane == 0) u[row] = acc;
}

// v[c] = sum_r W[r,c] * u[r]   (coalesced over columns)
__global__ __launch_bounds__(256) void mv_cols(
    const float* __restrict__ W, const float* __restrict__ u,
    float* __restrict__ v, int R, int C)
{
  const int c = blockIdx.x * 256 + threadIdx.x;
  if (c >= C) return;
  float acc = 0.f;
  for (int r = 0; r < R; ++r) acc += W[(size_t)r * C + c] * u[r];
  v[c] = acc;
}

__global__ __launch_bounds__(256) void normalize_vec(float* __restrict__ x, int n) {
  __shared__ float red[256];
  __shared__ float sc_s;
  float s = 0.f;
  for (int i = threadIdx.x; i < n; i += 256) { const float t = x[i]; s += t * t; }
  red[threadIdx.x] = s; __syncthreads();
  for (int o = 128; o > 0; o >>= 1) {
    if (threadIdx.x < o) red[threadIdx.x] += red[threadIdx.x + o];
    __syncthreads();
  }
  if (threadIdx.x == 0) sc_s = 1.0f / (sqrtf(red[0]) + 1e-12f);
  __syncthreads();
  const float sc = sc_s;
  for (int i = threadIdx.x; i < n; i += 256) x[i] *= sc;
}

__global__ __launch_bounds__(256) void dot_kernel(
    const float* __restrict__ a, const float* __restrict__ b,
    float* __restrict__ out, int n)
{
  __shared__ float red[256];
  float s = 0.f;
  for (int i = threadIdx.x; i < n; i += 256) s += a[i] * b[i];
  red[threadIdx.x] = s; __syncthreads();
  for (int o = 128; o > 0; o >>= 1) {
    if (threadIdx.x < o) red[threadIdx.x] += red[threadIdx.x + o];
    __syncthreads();
  }
  if (threadIdx.x == 0) out[0] = red[0];
}

// Wbf[i] = bf16(W[i] / sigma)
__global__ __launch_bounds__(256) void scale_cvt_kernel(
    const float* __restrict__ W, const float* __restrict__ sigma,
    bf16* __restrict__ out, size_t n)
{
  const size_t i = (size_t)blockIdx.x * 256 + threadIdx.x;
  if (i < n) out[i] = (bf16)(W[i] * (1.0f / sigma[0]));
}

__global__ __launch_bounds__(256) void cvt_kernel(
    const float* __restrict__ in, bf16* __restrict__ out, size_t n)
{
  const size_t i = (size_t)blockIdx.x * 256 + threadIdx.x;
  if (i < n) out[i] = (bf16)in[i];
}

// ---------------------------------------------------------------------------
// Host-side orchestration
// ---------------------------------------------------------------------------
extern "C" void kernel_launch(void* const* d_in, const int* in_sizes, int n_in,
                              void* d_out, int out_size, void* d_ws, size_t ws_size,
                              hipStream_t stream)
{
  const float* x       = (const float*)d_in[0];
  const float* embed_w = (const float*)d_in[1];
  const float* embed_b = (const float*)d_in[2];
  const float* W1      = (const float*)d_in[3];
  const float* b1      = (const float*)d_in[4];
  const float* W2      = (const float*)d_in[5];
  const float* b2      = (const float*)d_in[6];
  const float* ln_g    = (const float*)d_in[7];
  const float* ln_b    = (const float*)d_in[8];
  const float* head_w  = (const float*)d_in[9];
  const float* head_b  = (const float*)d_in[10];
  float* out = (float*)d_out;
  (void)in_sizes; (void)n_in; (void)out_size; (void)ws_size;

  char* ws = (char*)d_ws;
  size_t off = 0;
  auto alloc = [&](size_t bytes) -> char* {
    char* p = ws + off;
    off += (bytes + 255) & ~(size_t)255;
    return p;
  };

  float* u_v   = (float*)alloc(4096 * sizeof(float));
  float* v_v   = (float*)alloc(4096 * sizeof(float));
  float* t_v   = (float*)alloc(4096 * sizeof(float));
  float* sig   = (float*)alloc(8 * sizeof(float));
  bf16*  W1bf  = (bf16*)alloc((size_t)DFF * DH * sizeof(bf16));
  bf16*  W2bf  = (bf16*)alloc((size_t)DH * DFF * sizeof(bf16));
  bf16*  xbf   = (bf16*)alloc((size_t)BDIM * DIN * sizeof(bf16));
  bf16*  ewbf  = (bf16*)alloc((size_t)DH * DIN * sizeof(bf16));
  bf16*  hwbf  = (bf16*)alloc((size_t)DOUT * DH * sizeof(bf16));
  float* xemb  = (float*)alloc((size_t)BDIM * DH * sizeof(float));
  float* h     = (float*)alloc((size_t)BDIM * DH * sizeof(float));
  bf16*  hnbf  = (bf16*)alloc((size_t)BDIM * DH * sizeof(bf16));
  bf16*  hidbf = (bf16*)alloc((size_t)BDIM * DFF * sizeof(bf16));
  bf16*  hbf   = (bf16*)alloc((size_t)BDIM * DH * sizeof(bf16));

  const dim3 blk(256);

  // ---- spectral normalization of W (R x C), result -> bf16(W/sigma)
  auto spectral = [&](const float* W, int R, int C, bf16* Wbf) {
    fill_kernel<<<(C + 255) / 256, blk, 0, stream>>>(v_v, 1.0f / sqrtf((float)C), C);
    for (int it = 0; it < 15; ++it) {
      mv_rows<<<R / 8, blk, 0, stream>>>(W, v_v, u_v, R, C);
      normalize_vec<<<1, blk, 0, stream>>>(u_v, R);
      mv_cols<<<(C + 255) / 256, blk, 0, stream>>>(W, u_v, v_v, R, C);
      normalize_vec<<<1, blk, 0, stream>>>(v_v, C);
    }
    mv_rows<<<R / 8, blk, 0, stream>>>(W, v_v, t_v, R, C);
    dot_kernel<<<1, blk, 0, stream>>>(u_v, t_v, sig, R);
    const size_t n = (size_t)R * C;
    scale_cvt_kernel<<<(unsigned)((n + 255) / 256), blk, 0, stream>>>(W, sig, Wbf, n);
  };
  spectral(W1, DFF, DH, W1bf);
  spectral(W2, DH, DFF, W2bf);

  // ---- one-time bf16 converts
  {
    size_t n;
    n = (size_t)BDIM * DIN; cvt_kernel<<<(unsigned)((n + 255) / 256), blk, 0, stream>>>(x, xbf, n);
    n = (size_t)DH * DIN;   cvt_kernel<<<(unsigned)((n + 255) / 256), blk, 0, stream>>>(embed_w, ewbf, n);
    n = (size_t)DOUT * DH;  cvt_kernel<<<(unsigned)((n + 255) / 256), blk, 0, stream>>>(head_w, hwbf, n);
  }

  // ---- x_emb = x @ embed_w^T + embed_b ; h = x_emb
  {
    dim3 grid(DH / BN, BDIM / BM);
    gemm_bt<EPI_BIAS_F32><<<grid, blk, 0, stream>>>(
        xbf, ewbf, embed_b, xemb, nullptr, nullptr, nullptr, BDIM, DH, DIN);
  }
  hipMemcpyAsync(h, xemb, (size_t)BDIM * DH * sizeof(float),
                 hipMemcpyDeviceToDevice, stream);

  // ---- 30 fixed-point steps
  for (int s = 0; s < STEPS; ++s) {
    ln_bf16_kernel<<<BDIM, blk, 0, stream>>>(h, ln_g, ln_b, hnbf, DH);
    {
      dim3 grid(DFF / BN, BDIM / BM);  // hid = tanh(hn @ W1n^T + b1)
      gemm_bt<EPI_TANH_BF16><<<grid, blk, 0, stream>>>(
          hnbf, W1bf, b1, nullptr, hidbf, nullptr, nullptr, BDIM, DFF, DH);
    }
    {
      dim3 grid(DH / BN, BDIM / BM);   // h = (1-g)h + g(hid @ W2n^T + b2 + x_emb)
      gemm_bt<EPI_HUPDATE><<<grid, blk, 0, stream>>>(
          hidbf, W2bf, b2, nullptr, nullptr, xemb, h, BDIM, DH, DFF);
    }
  }

  // ---- head: out = h @ head_w^T + head_b
  {
    const size_t n = (size_t)BDIM * DH;
    cvt_kernel<<<(unsigned)((n + 255) / 256), blk, 0, stream>>>(h, hbf, n);
    dim3 grid(DOUT / BN, BDIM / BM);
    gemm_bt<EPI_BIAS_F32><<<grid, blk, 0, stream>>>(
        hbf, hwbf, head_b, out, nullptr, nullptr, nullptr, BDIM, DOUT, DH);
  }
}
